// GNeRF_15908558864501
// MI455X (gfx1250) — matrix operand order repaired and checked by reading, not compiled, not run on hardware
//
#include <hip/hip_runtime.h>
#include <hip/hip_bf16.h>
#include <cstdint>
#include <cstddef>

// ---------------------------------------------------------------------------
// NeRF forward (coarse + fine) for gfx1250: V_WMMA_F32_16X16X32_F16 matmuls,
// TENSOR_LOAD_TO_LDS (TDM) double-buffered weight staging.
// ---------------------------------------------------------------------------

typedef __attribute__((ext_vector_type(16))) _Float16 v16h;
typedef __attribute__((ext_vector_type(8)))  _Float16 v8h;
typedef __attribute__((ext_vector_type(8)))  float    v8f;
typedef __attribute__((ext_vector_type(4)))  unsigned v4u;
typedef __attribute__((ext_vector_type(8)))  int      v8i;
typedef __attribute__((ext_vector_type(4)))  int      v4i;

#define NRAYS 2048
#define NS_C  64
#define NS_F  128

#if __has_builtin(__builtin_amdgcn_tensor_load_to_lds)
#define HAVE_TDM 1
#endif

struct NerfOff {
    unsigned wt[8];
    unsigned wtfinal, wtsigma, wtrgb1, wtrgb2;
    unsigned b[8];
    unsigned bfinal, bsigma, brgb1, brgb2;
    unsigned edir;
};

// ---------------------------------------------------------------------------
// Weight prep: fp32 W[fin_real][fout_real] -> fp16 Wt[fout_pad][fin_pad]
// (transposed, zero-padded; optional zero-row insertion for skip concat).
// ---------------------------------------------------------------------------
__global__ void prep_w_kernel(const float* __restrict__ W, _Float16* __restrict__ dst,
                              int fin_real, int fout_real, int fin_pad, int fout_pad,
                              int insert_at) {
    int idx = blockIdx.x * 256 + threadIdx.x;
    if (idx >= fin_pad * fout_pad) return;
    int n = idx / fin_pad;
    int k = idx - n * fin_pad;
    float v = 0.f;
    int kk = k;
    bool zero = false;
    if (insert_at >= 0) {
        if (k == insert_at) zero = true;
        else if (k > insert_at) kk = k - 1;
    }
    if (!zero && kk < fin_real && n < fout_real) v = W[(size_t)kk * fout_real + n];
    dst[(size_t)n * fin_pad + k] = (_Float16)v;
}

__global__ void prep_b_kernel(const float* __restrict__ b, float* __restrict__ dst,
                              int n_real, int n_pad) {
    int idx = blockIdx.x * 256 + threadIdx.x;
    if (idx >= n_pad) return;
    dst[idx] = (idx < n_real) ? b[idx] : 0.f;
}

// ---------------------------------------------------------------------------
// Direction embedding: per ray 27 values (d, sin/cos of 4 freqs), padded to 32.
// ---------------------------------------------------------------------------
__global__ void embed_dir_kernel(const float* __restrict__ rays, _Float16* __restrict__ ed) {
    int idx = blockIdx.x * 256 + threadIdx.x;
    if (idx >= NRAYS * 32) return;
    int r = idx >> 5, col = idx & 31;
    float v = 0.f;
    if (col < 3) {
        v = rays[r * 10 + 3 + col];
    } else if (col < 27) {
        int j = col - 3;
        int f = j / 6, rem = j - f * 6;
        int dim = rem % 3, trig = rem / 3;
        float x = rays[r * 10 + 3 + dim] * exp2f((float)f);
        v = trig ? __cosf(x) : __sinf(x);
    }
    ed[idx] = (_Float16)v;
}

// Coarse z samples: z = near*(1-t)+far*t, t = s/63.
__global__ void zcoarse_kernel(const float* __restrict__ rays, float* __restrict__ zc) {
    int idx = blockIdx.x * 256 + threadIdx.x;
    if (idx >= NRAYS * NS_C) return;
    int r = idx >> 6, s = idx & 63;
    float t = (float)s * (1.0f / 63.0f);
    float nr = rays[r * 10 + 6], fr = rays[r * 10 + 7];
    zc[idx] = nr * (1.0f - t) + fr * t;
}

// ---------------------------------------------------------------------------
// TDM staging of one K-block (32 x FOUT fp16, row stride FIN in global) into
// LDS laid out as [FOUT][32].  One wave issues one TENSOR_LOAD_TO_LDS; other
// waves just hit the barrier.  Fallback: per-lane vector copy.
// ---------------------------------------------------------------------------
template<int FIN, int FOUT>
__device__ inline void stage_issue(const _Float16* __restrict__ Wt, int kb,
                                   _Float16* buf, int tid, int wv) {
#ifdef HAVE_TDM
    if (wv == 0) {
        unsigned long long ga = (unsigned long long)(uintptr_t)(Wt + (size_t)kb * 32);
        unsigned la = (unsigned)(uintptr_t)buf;  // generic->LDS offset (low 32 bits)
        // D# group 0: count=1 | lds_addr | global_addr[56:0] | type=2
        v4u g0 = { 1u, la, (unsigned)ga,
                   (unsigned)((ga >> 32) & 0x1FFFFFFu) | (2u << 30) };
        // D# group 1: data_size=2B; tensor_dim0=FIN, tensor_dim1=FOUT;
        // tile_dim0=32, tile_dim1=FOUT; tensor_dim0_stride=FIN.
        v8i g1 = { (int)(1u << 16),
                   (int)(((unsigned)FIN & 0xFFFFu) << 16),
                   (int)((((unsigned)FIN >> 16) & 0xFFFFu) | (((unsigned)FOUT & 0xFFFFu) << 16)),
                   (int)((((unsigned)FOUT >> 16) & 0xFFFFu) | (32u << 16)),
                   (int)((unsigned)FOUT & 0xFFFFu),
                   (int)(unsigned)FIN,
                   0, 0 };
        v4i g2 = { 0, 0, 0, 0 };
        v4i g3 = { 0, 0, 0, 0 };
        v8i g4 = { 0, 0, 0, 0, 0, 0, 0, 0 };
        __builtin_amdgcn_tensor_load_to_lds(g0, g1, g2, g3, g4, 0);
    }
#else
    if (FOUT == 256 || tid < FOUT) {
        const uint4* src = (const uint4*)(Wt + (size_t)tid * FIN + kb * 32);
        uint4* dst = (uint4*)(buf + tid * 32);
#pragma unroll
        for (int q = 0; q < 4; ++q) dst[q] = src[q];
    }
#endif
    (void)tid; (void)wv;
}

__device__ inline void stage_wait(int wv) {
#ifdef HAVE_TDM
    if (wv == 0) __builtin_amdgcn_s_wait_tensorcnt(0);
#endif
    (void)wv;
}

// ---------------------------------------------------------------------------
// One GEMM layer for a 32-point tile:  Out[32 x FOUT] = act(A[32 x FIN]*W + b)
// A from one or two LDS buffers (concat along K at C0).  Weights staged by
// TDM into double-buffered LDS; wave w: M-tile = w&1, N-tiles (w>>1)*4..+4.
// ACT: 0 relu, 1 none, 2 sigmoid.  GMODE: 0 none, 1 sigma, 2 rgb.
// ---------------------------------------------------------------------------
template<int S0, int C0, int S1, int FIN, int FOUT, int ACT, int GMODE>
__device__ inline void gemm_layer(
    const _Float16* A0, const _Float16* A1,
    const _Float16* __restrict__ Wt,
    const float* __restrict__ bias,
    _Float16* Hout, float* gout, int pbase,
    _Float16* W0, _Float16* W1)
{
    const int tid  = threadIdx.x;
    const int lane = tid & 31;
    const int wv   = __builtin_amdgcn_readfirstlane(tid) >> 5;   // uniform wave id
    const int hf   = lane >> 4;
    const int mn   = lane & 15;
    const int mtile = wv & 1;
    const int ngrp  = (wv >> 1) & 3;

    const v8f vzero = {0.f, 0.f, 0.f, 0.f, 0.f, 0.f, 0.f, 0.f};
    v8f acc[4];
#pragma unroll
    for (int j = 0; j < 4; ++j) acc[j] = vzero;

    constexpr int KB = FIN >> 5;

    __syncthreads();                       // prior layer done with W buffers
    stage_issue<FIN, FOUT>(Wt, 0, W0, tid, wv);

#pragma unroll
    for (int kb = 0; kb < KB; ++kb) {
        _Float16* cur = (kb & 1) ? W1 : W0;
        _Float16* nxt = (kb & 1) ? W0 : W1;
        stage_wait(wv);                    // wave 0: tensorcnt drained
        __syncthreads();                   // block sees staged K-block
        if (kb + 1 < KB) stage_issue<FIN, FOUT>(Wt, kb + 1, nxt, tid, wv);

        // A fragment (16x32 f16, ISA layout): slot s -> k = (s>>3)*16 + hf*8 + (s&7)
        int kl = kb * 32;
        const _Float16* src; int ss;
        if (kl < C0) { src = A0; ss = S0; } else { src = A1; ss = S1; kl -= C0; }
        const _Float16* ap = src + (mtile * 16 + mn) * ss + kl + hf * 8;
        v8h x0 = *(const v8h*)ap;
        v8h x1 = *(const v8h*)(ap + 16);
        v16h a = __builtin_shufflevector(x0, x1, 0, 1, 2, 3, 4, 5, 6, 7,
                                                 8, 9, 10, 11, 12, 13, 14, 15);
#pragma unroll
        for (int j = 0; j < 4; ++j) {
            int nt = ngrp * 4 + j;         // <=15, scalar; folds for FOUT==256
            if (nt * 16 < FOUT) {
                // B fragment (32x16 f16): lane n, slot s -> k = (s>>3)*16 + hf*8 + (s&7)
                const _Float16* bp = cur + (nt * 16 + mn) * 32 + hf * 8;
                v8h y0 = *(const v8h*)bp;
                v8h y1 = *(const v8h*)(bp + 16);
                v16h b = __builtin_shufflevector(y0, y1, 0, 1, 2, 3, 4, 5, 6, 7,
                                                         8, 9, 10, 11, 12, 13, 14, 15);
                acc[j] = __builtin_amdgcn_wmma_f32_16x16x32_f16(
                    false, a, false, b, (short)0, acc[j], false, false);
            }
        }
    }

    // Epilogue: bias + activation; D layout: vgpr r, lane l -> m=r+8*hf, n=l&15
#pragma unroll
    for (int j = 0; j < 4; ++j) {
        int nt = ngrp * 4 + j;
        if (nt * 16 >= FOUT) continue;
        int ncol = nt * 16 + mn;
        float bv = bias[ncol];
#pragma unroll
        for (int r = 0; r < 8; ++r) {
            int row = mtile * 16 + hf * 8 + r;
            float v = acc[j][r] + bv;
            if (ACT == 0)      v = fmaxf(v, 0.f);
            else if (ACT == 2) v = 1.f / (1.f + __expf(-v));
            if (Hout) Hout[row * 256 + ncol] = (_Float16)v;
            if (GMODE == 1 && ncol == 0) gout[(size_t)(pbase + row) * 4 + 3] = v;
            if (GMODE == 2 && ncol < 3)  gout[(size_t)(pbase + row) * 4 + ncol] = v;
        }
    }
}

// ---------------------------------------------------------------------------
// Fused NeRF MLP: 32 points per block, embedding computed in-kernel.
// Writes rgbsigma[point] = {r,g,b,sigma_raw}.
// ---------------------------------------------------------------------------
__global__ __launch_bounds__(256) void nerf_mlp_kernel(
    const float* __restrict__ rays, const char* __restrict__ ws,
    const float* __restrict__ zbuf, int ns,
    float* __restrict__ rgbsigma, NerfOff P)
{
    __shared__ __attribute__((aligned(32))) _Float16 E[32 * 64];
    __shared__ __attribute__((aligned(32))) _Float16 H[32 * 256];
    __shared__ __attribute__((aligned(32))) _Float16 H2[32 * 256];
    __shared__ __attribute__((aligned(32))) _Float16 Dd[32 * 32];
    __shared__ __attribute__((aligned(32))) _Float16 Wlds[2][256 * 32];

    const int pbase = blockIdx.x * 32;

    // Positional embedding -> E (32 rows x 64 cols, col 63 = 0 pad)
    for (int i = threadIdx.x; i < 32 * 64; i += 256) {
        int row = i >> 6, col = i & 63;
        int pt = pbase + row;
        int r = pt / ns;
        float z = zbuf[pt];
        float val = 0.f;
        if (col < 3) {
            val = rays[r * 10 + col] + rays[r * 10 + 3 + col] * z;
        } else if (col < 63) {
            int j = col - 3;
            int f = j / 6, rem = j - f * 6;
            int dim = rem % 3, trig = rem / 3;
            float x = (rays[r * 10 + dim] + rays[r * 10 + 3 + dim] * z) * exp2f((float)f);
            val = trig ? __cosf(x) : __sinf(x);
        }
        E[i] = (_Float16)val;
    }
    // Direction embedding tile -> Dd (32 rows x 32 cols)
    const _Float16* edir = (const _Float16*)(ws + P.edir);
    for (int i = threadIdx.x; i < 32 * 32; i += 256) {
        int row = i >> 5, col = i & 31;
        int r = (pbase + row) / ns;
        Dd[i] = edir[r * 32 + col];
    }

    const char* w = ws;
    #define WT(o) ((const _Float16*)(w + (o)))
    #define BF(o) ((const float*)(w + (o)))

    // 8 hidden layers (skip concat at layer 4), ping-pong H <-> H2.
    gemm_layer< 64, 64,   0,  64, 256, 0, 0>(E,  nullptr, WT(P.wt[0]), BF(P.b[0]), H,  nullptr, pbase, Wlds[0], Wlds[1]);
    gemm_layer<256, 256,  0, 256, 256, 0, 0>(H,  nullptr, WT(P.wt[1]), BF(P.b[1]), H2, nullptr, pbase, Wlds[0], Wlds[1]);
    gemm_layer<256, 256,  0, 256, 256, 0, 0>(H2, nullptr, WT(P.wt[2]), BF(P.b[2]), H,  nullptr, pbase, Wlds[0], Wlds[1]);
    gemm_layer<256, 256,  0, 256, 256, 0, 0>(H,  nullptr, WT(P.wt[3]), BF(P.b[3]), H2, nullptr, pbase, Wlds[0], Wlds[1]);
    gemm_layer< 64, 64, 256, 320, 256, 0, 0>(E,  H2,      WT(P.wt[4]), BF(P.b[4]), H,  nullptr, pbase, Wlds[0], Wlds[1]);
    gemm_layer<256, 256,  0, 256, 256, 0, 0>(H,  nullptr, WT(P.wt[5]), BF(P.b[5]), H2, nullptr, pbase, Wlds[0], Wlds[1]);
    gemm_layer<256, 256,  0, 256, 256, 0, 0>(H2, nullptr, WT(P.wt[6]), BF(P.b[6]), H,  nullptr, pbase, Wlds[0], Wlds[1]);
    gemm_layer<256, 256,  0, 256, 256, 0, 0>(H,  nullptr, WT(P.wt[7]), BF(P.b[7]), H2, nullptr, pbase, Wlds[0], Wlds[1]);
    // sigma head (raw; relu applied at compositing)
    gemm_layer<256, 256,  0, 256,  16, 1, 1>(H2, nullptr, WT(P.wtsigma), BF(P.bsigma), nullptr, rgbsigma, pbase, Wlds[0], Wlds[1]);
    // feature head -> H (no activation)
    gemm_layer<256, 256,  0, 256, 256, 1, 0>(H2, nullptr, WT(P.wtfinal), BF(P.bfinal), H, nullptr, pbase, Wlds[0], Wlds[1]);
    // rgb1: concat(feat, dir) -> 128, relu
    gemm_layer<256, 256, 32, 288, 128, 0, 0>(H,  Dd,      WT(P.wtrgb1), BF(P.brgb1), H2, nullptr, pbase, Wlds[0], Wlds[1]);
    // rgb2: 128 -> 3, sigmoid, write rgb
    gemm_layer<256, 256,  0, 128,  16, 2, 2>(H2, nullptr, WT(P.wtrgb2), BF(P.brgb2), nullptr, rgbsigma, pbase, Wlds[0], Wlds[1]);
    #undef WT
    #undef BF
}

// ---------------------------------------------------------------------------
// Volumetric compositing (1 thread per ray).  out = rgb*2-1.  Optionally store
// per-sample weights for importance sampling.
// ---------------------------------------------------------------------------
__global__ void composite_kernel(const float* __restrict__ rays,
                                 const float* __restrict__ zbuf, int ns,
                                 const float* __restrict__ rgbsigma,
                                 float* __restrict__ outrgb,
                                 float* __restrict__ wstore)
{
    int r = blockIdx.x * 256 + threadIdx.x;
    if (r >= NRAYS) return;
    float dx = rays[r * 10 + 3], dy = rays[r * 10 + 4], dz = rays[r * 10 + 5];
    float dn = sqrtf(dx * dx + dy * dy + dz * dz);
    float fr = rays[r * 10 + 7];
    float T = 1.f, cr = 0.f, cg = 0.f, cb = 0.f;
    float zc = zbuf[(size_t)r * ns];
    for (int s = 0; s < ns; ++s) {
        float zn = (s + 1 < ns) ? zbuf[(size_t)r * ns + s + 1] : fr;
        float delta = (zn - zc) * dn;
        const float* p = rgbsigma + (size_t)(r * ns + s) * 4;
        float sg = fmaxf(p[3], 0.f);
        float a = 1.f - __expf(-delta * sg);
        float wgt = a * T;
        cr += wgt * p[0]; cg += wgt * p[1]; cb += wgt * p[2];
        if (wstore) wstore[(size_t)r * ns + s] = wgt;
        T *= (1.f - a + 1e-10f);
        zc = zn;
    }
    outrgb[r * 3 + 0] = cr * 2.f - 1.f;
    outrgb[r * 3 + 1] = cg * 2.f - 1.f;
    outrgb[r * 3 + 2] = cb * 2.f - 1.f;
}

// ---------------------------------------------------------------------------
// Inverse-CDF importance sampling + merge with coarse z (1 thread per ray).
// ---------------------------------------------------------------------------
__global__ void pdf_kernel(const float* __restrict__ zc,
                           const float* __restrict__ wc,
                           float* __restrict__ zf)
{
    int r = blockIdx.x * 256 + threadIdx.x;
    if (r >= NRAYS) return;
    float cdf[63], bins[63], nz[64];
    float sum = 0.f;
    for (int i = 0; i < 62; ++i) sum += wc[(size_t)r * NS_C + 1 + i] + 1e-5f;
    float inv = 1.f / sum;
    cdf[0] = 0.f;
    float c = 0.f;
    for (int i = 0; i < 62; ++i) {
        c += (wc[(size_t)r * NS_C + 1 + i] + 1e-5f) * inv;
        cdf[i + 1] = c;
    }
    for (int i = 0; i < 63; ++i)
        bins[i] = 0.5f * (zc[(size_t)r * NS_C + i] + zc[(size_t)r * NS_C + i + 1]);
    for (int j = 0; j < 64; ++j) {
        float u = (float)j * (1.f / 63.f);
        int ind = 0;
        for (int i = 0; i < 63; ++i)
            if (cdf[i] <= u) ind = i + 1;     // searchsorted right
        int below = ind - 1; if (below < 0) below = 0;
        int above = ind;     if (above > 62) above = 62;
        float cg0 = cdf[below], cg1 = cdf[above];
        float bg0 = bins[below], bg1 = bins[above];
        float den = (cg1 - cg0 < 1e-5f) ? 1.f : (cg1 - cg0);
        nz[j] = bg0 + (u - cg0) / den * (bg1 - bg0);
    }
    int i = 0, j = 0;
    for (int k = 0; k < NS_F; ++k) {
        float a = (i < 64) ? zc[(size_t)r * NS_C + i] : 3.4e38f;
        float b = (j < 64) ? nz[j] : 3.4e38f;
        if (a <= b) { zf[(size_t)r * NS_F + k] = a; ++i; }
        else        { zf[(size_t)r * NS_F + k] = b; ++j; }
    }
}

// ---------------------------------------------------------------------------
// Host driver
// ---------------------------------------------------------------------------
extern "C" void kernel_launch(void* const* d_in, const int* in_sizes, int n_in,
                              void* d_out, int out_size, void* d_ws, size_t ws_size,
                              hipStream_t stream)
{
    (void)out_size; (void)ws_size;
    const float* rays = (const float*)d_in[0];

    // Param index maps for the two plausible pytree flattenings.
    int iW[12], iB[12]; // 0..7 = xyz layers, 8=final, 9=sigma, 10=rgb1, 11=rgb2
    bool insertion = (n_in > 1 && in_sizes[1] == 63 * 256);
    if (insertion) {
        for (int l = 0; l < 8; ++l) { iW[l] = 1 + 2 * l; iB[l] = 2 + 2 * l; }
        iW[8] = 17; iB[8] = 18;   // final
        iW[9] = 19; iB[9] = 20;   // sigma
        iW[10] = 21; iB[10] = 22; // rgb1
        iW[11] = 23; iB[11] = 24; // rgb2
    } else {
        iW[8] = 1;  iB[8] = 2;    // final
        iW[10] = 3; iB[10] = 4;   // rgb1
        iW[11] = 5; iB[11] = 6;   // rgb2
        iW[9] = 7;  iB[9] = 8;    // sigma
        for (int l = 0; l < 8; ++l) { iW[l] = 9 + 2 * l; iB[l] = 10 + 2 * l; }
    }

    // Workspace layout (all offsets 256B aligned)
    size_t off = 0;
    auto alloc = [&](size_t bytes) { size_t o = off; off += (bytes + 255) & ~(size_t)255; return o; };
    NerfOff P;
    const int finp_l[8] = {64, 256, 256, 256, 320, 256, 256, 256};
    for (int l = 0; l < 8; ++l) P.wt[l] = (unsigned)alloc((size_t)finp_l[l] * 256 * 2);
    P.wtfinal = (unsigned)alloc(256 * 256 * 2);
    P.wtsigma = (unsigned)alloc(256 * 16 * 2);
    P.wtrgb1  = (unsigned)alloc(288 * 128 * 2);
    P.wtrgb2  = (unsigned)alloc(128 * 16 * 2);
    for (int l = 0; l < 8; ++l) P.b[l] = (unsigned)alloc(256 * 4);
    P.bfinal = (unsigned)alloc(256 * 4);
    P.bsigma = (unsigned)alloc(16 * 4);
    P.brgb1  = (unsigned)alloc(128 * 4);
    P.brgb2  = (unsigned)alloc(16 * 4);
    P.edir   = (unsigned)alloc((size_t)NRAYS * 32 * 2);
    size_t o_zc  = alloc((size_t)NRAYS * NS_C * 4);
    size_t o_zf  = alloc((size_t)NRAYS * NS_F * 4);
    size_t o_rsC = alloc((size_t)NRAYS * NS_C * 16);
    size_t o_rsF = alloc((size_t)NRAYS * NS_F * 16);
    size_t o_wC  = alloc((size_t)NRAYS * NS_C * 4);

    char* ws = (char*)d_ws;

    // Weight/bias prep
    struct Spec { int wi, bi; unsigned wo, bo; int finr, foutr, finp, foutp, ins_at; };
    Spec sp[12];
    for (int l = 0; l < 8; ++l) {
        int finr = (l == 0) ? 63 : (l == 4) ? 319 : 256;
        sp[l] = { iW[l], iB[l], P.wt[l], P.b[l], finr, 256, finp_l[l], 256, (l == 4) ? 63 : -1 };
    }
    sp[8]  = { iW[8],  iB[8],  P.wtfinal, P.bfinal, 256, 256, 256, 256, -1 };
    sp[9]  = { iW[9],  iB[9],  P.wtsigma, P.bsigma, 256, 1,   256, 16,  -1 };
    sp[10] = { iW[10], iB[10], P.wtrgb1,  P.brgb1,  283, 128, 288, 128, -1 };
    sp[11] = { iW[11], iB[11], P.wtrgb2,  P.brgb2,  128, 3,   128, 16,  -1 };
    for (int s = 0; s < 12; ++s) {
        int tot = sp[s].finp * sp[s].foutp;
        prep_w_kernel<<<(tot + 255) / 256, 256, 0, stream>>>(
            (const float*)d_in[sp[s].wi], (_Float16*)(ws + sp[s].wo),
            sp[s].finr, sp[s].foutr, sp[s].finp, sp[s].foutp, sp[s].ins_at);
        prep_b_kernel<<<1, 256, 0, stream>>>(
            (const float*)d_in[sp[s].bi], (float*)(ws + sp[s].bo),
            sp[s].foutr, sp[s].foutp);
    }

    embed_dir_kernel<<<(NRAYS * 32) / 256, 256, 0, stream>>>(rays, (_Float16*)(ws + P.edir));
    zcoarse_kernel<<<(NRAYS * NS_C) / 256, 256, 0, stream>>>(rays, (float*)(ws + o_zc));

    // Coarse pass: 131072 points / 32 per block
    nerf_mlp_kernel<<<(NRAYS * NS_C) / 32, 256, 0, stream>>>(
        rays, ws, (const float*)(ws + o_zc), NS_C, (float*)(ws + o_rsC), P);
    composite_kernel<<<NRAYS / 256, 256, 0, stream>>>(
        rays, (const float*)(ws + o_zc), NS_C, (const float*)(ws + o_rsC),
        (float*)d_out, (float*)(ws + o_wC));

    // Importance sampling -> z_fine (sorted merge of coarse z and new samples)
    pdf_kernel<<<NRAYS / 256, 256, 0, stream>>>(
        (const float*)(ws + o_zc), (const float*)(ws + o_wC), (float*)(ws + o_zf));

    // Fine pass: 262144 points / 32 per block
    nerf_mlp_kernel<<<(NRAYS * NS_F) / 32, 256, 0, stream>>>(
        rays, ws, (const float*)(ws + o_zf), NS_F, (float*)(ws + o_rsF), P);
    composite_kernel<<<NRAYS / 256, 256, 0, stream>>>(
        rays, (const float*)(ws + o_zf), NS_F, (const float*)(ws + o_rsF),
        (float*)d_out + (size_t)NRAYS * 3, nullptr);
}
